// RNN_fly_2095944040500
// MI455X (gfx1250) — matrix operand order repaired and checked by reading, not compiled, not run on hardware
//
#include <hip/hip_runtime.h>
#include <hip/hip_bf16.h>

typedef __attribute__((ext_vector_type(16))) _Float16 v16h;
typedef __attribute__((ext_vector_type(8)))  float    v8f;

namespace {
constexpr int kB = 32, kT = 512, kI = 64, kH = 2048, kO = 16;
constexpr int kKtot = kH + kI;     // 2112 (fused K: recurrent + input proj)
constexpr int kNKT  = kKtot / 32;  // 66 k-tiles for the main GEMM
constexpr int kNKTH = kH / 32;     // 64 k-tiles for the readout GEMM
constexpr int kAbuf = kNKT * 512;  // halves per A staging buffer
}

__device__ __forceinline__ float softplus_f(float x) {
  return (x > 20.0f) ? x : log1pf(__expf(x));
}

// Swizzled half-index of A-matrix element (row m, col k) in the wave32
// 16x32 f16 WMMA A layout (ISA 7.12.2): per 32-wide k-tile,
//   lanes 0-15  : row m, K {0..7, 16..23}
//   lanes 16-31 : row m, K {8..15, 24..31}
__device__ __forceinline__ int a_swz(int m, int k) {
  int kt = k >> 5, kc = k & 31, grp = kc >> 3;
  int lane = m + ((grp & 1) << 4);
  int half = (kc & 7) + ((grp >> 1) << 3);
  return (kt << 9) + (lane << 4) + half;
}

// ---------------- prep: fuse + transpose + swizzle weights to f16 ----------
// W[k][n] (k<2048 : Jmat^T = exp(wrec[n][k])*mwrec[n][k];
//          k>=2048: win    = exp(si[k-2048])*wi[k-2048][n])
// stored per 32x16 (KxN) tile in the wave32 B-fragment layout:
//   lane = (n%16) + 16*(((k%32)/8)%2), half = k%8 + 8*((k%32)/16)  -> 16 halves/lane contiguous
__global__ void prep_w(const float* __restrict__ wrec, const float* __restrict__ mwrec,
                       const float* __restrict__ wi,   const float* __restrict__ si,
                       _Float16* __restrict__ Wswz) {
  int idx = blockIdx.x * 256 + threadIdx.x;
  if (idx >= kKtot * kH) return;
  int k = idx / kH, n = idx - k * kH;
  float v;
  if (k < kH) {
    v = __expf(wrec[n * kH + k]) * mwrec[n * kH + k];
  } else {
    int j = k - kH;
    v = __expf(si[j]) * wi[j * kH + n];
  }
  int kt = k >> 5, kc = k & 31, ntile = n >> 4, nc = n & 15;
  int lane = nc + (((kc >> 3) & 1) << 4);
  int half = (kc & 7) + ((kc >> 4) << 3);
  Wswz[(((size_t)(ntile * kNKT + kt)) << 9) + (lane << 4) + half] = (_Float16)v;
}

__global__ void prep_wout(const float* __restrict__ wout, _Float16* __restrict__ Wo) {
  int idx = blockIdx.x * 256 + threadIdx.x;
  if (idx >= kH * kO) return;
  int k = idx / kO, o = idx - k * kO;
  int kt = k >> 5, kc = k & 31;
  int lane = o + (((kc >> 3) & 1) << 4);
  int half = (kc & 7) + ((kc >> 4) << 3);
  Wo[((size_t)kt << 9) + (lane << 4) + half] = (_Float16)wout[k * kO + o];
}

// ---------------- persistent RNN scan -------------------------------------
// grid = 2 blocks (16 batch rows each), 1024 threads = 32 wave32s/WGP.
// wave w owns hidden columns [w*64, w*64+64): 4 WMMA C tiles, h kept in VGPRs.
// Double-buffered shared A holds [eg*r | x_t] (16 x 2112 f16) in WMMA A
// swizzle: step p reads buf[p&1], epilogue writes buf[(p+1)&1] -> ONE barrier
// per step orders writers ahead of next step's readers.
__global__ __launch_bounds__(1024)
void rnn_scan(const float* __restrict__ input, const float* __restrict__ noise,
              const float* __restrict__ g,     const float* __restrict__ b,
              const float* __restrict__ taus,  const float* __restrict__ h0,
              const _Float16* __restrict__ Wswz, const _Float16* __restrict__ Wo,
              float* __restrict__ out) {
  __shared__ __align__(32) _Float16 ldsAbuf[2 * kAbuf];  // 135 KB
  __shared__ float ldsEg[kH], ldsBb[kH], ldsIt[kH];      // 24 KB

  const int tid   = threadIdx.x;
  const int wave  = tid >> 5;
  const int lane  = tid & 31;
  const int b0    = blockIdx.x * 16;
  const int nrow  = lane & 15;   // N within C tile
  const int mhalf = lane >> 4;   // 0: rows 0-7, 1: rows 8-15

  // per-column parameters
  for (int n = tid; n < kH; n += 1024) {
    ldsEg[n] = __expf(g[n]);
    ldsBb[n] = b[n];
    ldsIt[n] = 1.0f / (2.6f + 2.4f * tanhf(taus[n]));  // 0.5*(5+.2), 0.5*(5-.2)
  }
  __syncthreads();

  // stage eg*r0 into buffer 0 (r0 = softplus(h0+b), same for all batch rows)
  for (int e = tid; e < 16 * kH; e += 1024) {
    int m = e >> 11, k = e & (kH - 1);
    float r0 = softplus_f(h0[k] + ldsBb[k]);
    ldsAbuf[a_swz(m, k)] = (_Float16)(ldsEg[k] * r0);
  }
  // stage x_0 into buffer 0
  {
    int m = tid >> 6, j = tid & 63;
    float xv = input[((size_t)(b0 + m) * kT + 0) * kI + j];
    ldsAbuf[a_swz(m, kH + j)] = (_Float16)xv;
  }

  // h state in registers, exact WMMA C layout
  v8f hreg[4];
#pragma unroll
  for (int nt = 0; nt < 4; ++nt) {
    float h0n = h0[(wave << 6) + (nt << 4) + nrow];
#pragma unroll
    for (int j = 0; j < 8; ++j) hreg[nt][j] = h0n;
  }
  __syncthreads();

  // per-wave base of this wave's B-tile column block (4 ntiles)
  const _Float16* __restrict__ Wbase =
      Wswz + (((size_t)(wave << 2) * kNKT) << 9) + (lane << 4);

  for (int p = 0; p < kT; ++p) {
    const bool do_step = (p < kT - 1);
    const _Float16* ldsA  = ldsAbuf + (p & 1) * kAbuf;         // read buffer
    _Float16*       ldsAn = ldsAbuf + ((p + 1) & 1) * kAbuf;   // write buffer
    v8f acc[4] = {};

    if (do_step) {
      // pre-activation GEMM: (eg*r | x_p) @ (Jmat^T ; win), K = 2112.
      // Load all 4 B tiles into distinct regs BEFORE the 4 WMMAs so the
      // waits stagger and L2 latency hides under the matrix ops.
      for (int kt = 0; kt < kNKT; ++kt) {
        v16h a = *(const v16h*)&ldsA[(kt << 9) + (lane << 4)];
        v16h bt[4];
#pragma unroll
        for (int nt = 0; nt < 4; ++nt)
          bt[nt] = *(const v16h*)(Wbase + (((size_t)nt * kNKT + kt) << 9));
        if (kt + 1 < kNKT)
          __builtin_prefetch(Wbase + (((size_t)(kt + 1)) << 9), 0, 2);
#pragma unroll
        for (int nt = 0; nt < 4; ++nt)
          acc[nt] = __builtin_amdgcn_wmma_f32_16x16x32_f16(
              false, a, false, bt[nt], (short)0, acc[nt], false, false);
      }
    }

    // readout for position p: (eg*r_p) @ wout — wave 0, K = 2048
    if (wave == 0) {
      v8f oacc = {};
      for (int kt = 0; kt < kNKTH; ++kt) {
        v16h a = *(const v16h*)&ldsA[(kt << 9) + (lane << 4)];
        const v16h bm = *(const v16h*)(Wo + ((size_t)kt << 9) + (lane << 4));
        oacc = __builtin_amdgcn_wmma_f32_16x16x32_f16(
            false, a, false, bm, (short)0, oacc, false, false);
      }
#pragma unroll
      for (int j = 0; j < 8; ++j) {
        int m = j + (mhalf << 3);
        out[((size_t)(b0 + m) * kT + p) * kO + nrow] = oacc[j];
      }
    }

    if (do_step) {
      // epilogue: h += noise_std*noi + alpha*inv_tau*(-h + gemm); r = softplus;
      // restage eg*r (and x_{p+1}) into the OTHER buffer — no read conflict.
#pragma unroll
      for (int nt = 0; nt < 4; ++nt) {
        int n = (wave << 6) + (nt << 4) + nrow;
        float egn = ldsEg[n], bbn = ldsBb[n], itn = ldsIt[n];
#pragma unroll
        for (int j = 0; j < 8; ++j) {
          int m = j + (mhalf << 3);
          float noi = noise[((size_t)(b0 + m) * kT + p) * kH + n];
          float hv = hreg[nt][j];
          hv = hv + 0.005f * noi + 0.2f * itn * (acc[nt][j] - hv);
          hreg[nt][j] = hv;
          float rv = softplus_f(hv + bbn);
          ldsAn[a_swz(m, n)] = (_Float16)(egn * rv);
        }
      }
      // stage x_{p+1}
      {
        int m = tid >> 6, j = tid & 63;
        float xv = input[((size_t)(b0 + m) * kT + (p + 1)) * kI + j];
        ldsAn[a_swz(m, kH + j)] = (_Float16)xv;
      }
    }
    __syncthreads();  // writers of buf[(p+1)&1] before next step's readers
  }
}

extern "C" void kernel_launch(void* const* d_in, const int* in_sizes, int n_in,
                              void* d_out, int out_size, void* d_ws, size_t ws_size,
                              hipStream_t stream) {
  (void)in_sizes; (void)n_in; (void)out_size; (void)ws_size;
  const float* input = (const float*)d_in[0];
  const float* noise = (const float*)d_in[1];
  const float* wi    = (const float*)d_in[2];
  const float* si    = (const float*)d_in[3];
  const float* wrec  = (const float*)d_in[4];
  const float* mwrec = (const float*)d_in[5];
  const float* wout  = (const float*)d_in[6];
  const float* b     = (const float*)d_in[7];
  const float* g     = (const float*)d_in[8];
  const float* taus  = (const float*)d_in[9];
  const float* h0    = (const float*)d_in[10];

  _Float16* Wswz = (_Float16*)d_ws;                   // 2112*2048 f16 = 8.25 MB
  _Float16* Wo   = Wswz + (size_t)kKtot * kH;         // 2048*16 f16

  prep_w<<<(kKtot * kH + 255) / 256, 256, 0, stream>>>(wrec, mwrec, wi, si, Wswz);
  prep_wout<<<(kH * kO + 255) / 256, 256, 0, stream>>>(wout, Wo);
  rnn_scan<<<kB / 16, 1024, 0, stream>>>(input, noise, g, b, taus, h0, Wswz, Wo,
                                         (float*)d_out);
}
